// Attention_12987981103117
// MI455X (gfx1250) — matrix operand order repaired and checked by reading, not compiled, not run on hardware
//
#include <hip/hip_runtime.h>
#include <hip/hip_fp16.h>

typedef __attribute__((ext_vector_type(16))) _Float16 v16h;
typedef __attribute__((ext_vector_type(8)))  _Float16 v8h;
typedef __attribute__((ext_vector_type(8)))  float    v8f;

#define B_   4
#define H_   16
#define S_   1024
#define D_   64
#define M_   64
#define BH_  (B_*H_)
#define EPSF 1e-4f
#define DN   0.3535533905932738f   /* 64^-0.25 */
#define DIAGC 0.0625f              /* 0.5 * 64^-0.5 */

// ---------------- WMMA fragment loaders (CDNA5 16x16x32 f16 layouts) ----------------
// A 16x32 f16: lane: row = lane&15, base0 = (lane>=16)?8:0.
//   V0-3 hold K = k0+base0 .. +7 ; V4-7 hold K = k0+base0+16 .. +23.
__device__ inline v16h load_afrag(const _Float16* src, int ld, int row, int k0, int half) {
  const _Float16* p = src + row * ld + k0 + half * 8;
  v8h x = *(const v8h*)(p);
  v8h y = *(const v8h*)(p + 16);
  v16h a;
#pragma unroll
  for (int i = 0; i < 8; ++i) { a[i] = x[i]; a[i + 8] = y[i]; }
  return a;
}
// B 32x16 f16 from transposed storage srcT[col][k]: lane: col = lane&15,
//   base = (lane>=16)?8:0. V0-3 hold K = k0+base .. +7 ; V4-7 hold K = k0+base+16 .. +23.
__device__ inline v16h load_bfrag(const _Float16* srcT, int ld, int col, int k0, int half) {
  const _Float16* p = srcT + col * ld + k0 + half * 8;
  v8h x = *(const v8h*)(p);
  v8h y = *(const v8h*)(p + 16);
  v16h b;
#pragma unroll
  for (int i = 0; i < 8; ++i) { b[i] = x[i]; b[i + 8] = y[i]; }
  return b;
}

__device__ inline v8f wmma16(v16h a, v16h b, v8f c) {
  return __builtin_amdgcn_wmma_f32_16x16x32_f16(false, a, false, b, (short)0, c, false, false);
}

// CDNA5 async global->LDS copy (ASYNCcnt-tracked).  VDST = LDS byte offset,
// VADDR = 64-bit global address (GV mode).
__device__ inline void async_b128(unsigned lds_off, const void* g) {
  asm volatile("global_load_async_to_lds_b128 %0, %1, off"
               :: "v"(lds_off), "v"(g) : "memory");
}

// ---------------- stage 1: convert / scale / transpose ----------------
__global__ void k_prep_qk(const float* __restrict__ q, const float* __restrict__ k,
                          _Float16* __restrict__ qh, _Float16* __restrict__ kh) {
  int i = blockIdx.x * 256 + threadIdx.x;          // BH_*S_*D_ / 256 blocks
  qh[i] = (_Float16)(q[i] * DN);
  kh[i] = (_Float16)(k[i] * DN);
}

__global__ void k_diag(const float* __restrict__ q, const float* __restrict__ k,
                       float* __restrict__ dq, float* __restrict__ dk) {
  int w    = (blockIdx.x * blockDim.x + threadIdx.x) >> 5;  // one wave per row
  int lane = threadIdx.x & 31;
  const float* qr = q + (size_t)w * 64;
  const float* kr = k + (size_t)w * 64;
  float a = qr[lane], b = qr[lane + 32];
  float sq = a * a + b * b;
  a = kr[lane]; b = kr[lane + 32];
  float sk = a * a + b * b;
#pragma unroll
  for (int off = 16; off > 0; off >>= 1) {
    sq += __shfl_xor(sq, off, 32);
    sk += __shfl_xor(sk, off, 32);
  }
  if (lane == 0) { dq[w] = sq * DIAGC; dk[w] = sk * DIAGC; }
}

__global__ void k_vt(const float* __restrict__ v, _Float16* __restrict__ vt) {
  // vt[b,h,d,s] = f16(v[b,h,s,d]); grid = BH_*(S_/64), 256 threads
  __shared__ _Float16 t[64][66];   // padded stride: 33 dwords -> spread banks
  int bh = blockIdx.x >> 4, st = blockIdx.x & 15;
  const float* src = v + ((size_t)bh * S_ + st * 64) * 64;
  for (int i = threadIdx.x; i < 4096; i += 256) {
    int s = i >> 6, d = i & 63;
    t[s][d] = (_Float16)src[i];
  }
  __syncthreads();
  _Float16* dst = vt + (size_t)bh * D_ * S_ + st * 64;
  for (int i = threadIdx.x; i < 4096; i += 256) {
    int d = i >> 6, s = i & 63;
    dst[(size_t)d * S_ + s] = t[s][d];
  }
}

__global__ void k_pt(const float* __restrict__ P, _Float16* __restrict__ Pt,
                     int* __restrict__ kmax) {
  for (int idx = threadIdx.x; idx < 4096; idx += 256) {
    int m = idx >> 6, d = idx & 63;
    Pt[idx] = (_Float16)P[d * 64 + m];   // Pt[m][d]
  }
  if (threadIdx.x < BH_) kmax[threadIdx.x] = (int)0x80000000;
}

// ---------------- projection core: one wave -> 16 rows x 64 feats ----------------
__device__ inline void proj_tiles(const _Float16* xrows, const _Float16* ptT,
                                  v8f c[4], int lane) {
  int row = lane & 15, half = lane >> 4;
  v16h a0 = load_afrag(xrows, 64, row, 0, half);
  v16h a1 = load_afrag(xrows, 64, row, 32, half);
#pragma unroll
  for (int n = 0; n < 4; ++n) {
    v8f acc = {};
    v16h b0 = load_bfrag(ptT, 64, n * 16 + row, 0, half);
    v16h b1 = load_bfrag(ptT, 64, n * 16 + row, 32, half);
    acc = wmma16(a0, b0, acc);
    acc = wmma16(a1, b1, acc);
    c[n] = acc;
  }
}

// ---------------- stage 2: kdash + per-head max ----------------
__global__ void __launch_bounds__(128)
k_kdash(const _Float16* __restrict__ kh, const _Float16* __restrict__ Pt_g,
        const float* __restrict__ dk, float* __restrict__ kdash,
        int* __restrict__ kmax) {
  __shared__ alignas(16) _Float16 pt[4096];
  __shared__ float ldiag[64];
  __shared__ float wmaxs[4];
  int bh = blockIdx.x >> 4, tile = blockIdx.x & 15;
  int tid = threadIdx.x, w = tid >> 5, lane = tid & 31;
  {
    const uint4* s = (const uint4*)Pt_g; uint4* d = (uint4*)pt;
    for (int i = tid; i < 512; i += 128) d[i] = s[i];
  }
  int rowbase = bh * S_ + tile * 64;
  if (tid < 64) ldiag[tid] = dk[rowbase + tid];
  __syncthreads();
  v8f c[4];
  proj_tiles(kh + (size_t)(rowbase + w * 16) * 64, pt, c, lane);
  int row = lane & 15, half = lane >> 4;
  float lm = -3.0e38f;
  float* ob = kdash + (size_t)(rowbase + w * 16) * 64;
#pragma unroll
  for (int n = 0; n < 4; ++n)
#pragma unroll
    for (int r = 0; r < 8; ++r) {
      float dash = c[n][r];
      lm = fmaxf(lm, dash);                               // max of data_dash
      int rr = half * 8 + r;
      ob[rr * 64 + n * 16 + row] = dash - ldiag[w * 16 + rr];
    }
#pragma unroll
  for (int off = 16; off > 0; off >>= 1) lm = fmaxf(lm, __shfl_xor(lm, off, 32));
  if (lane == 0) wmaxs[w] = lm;
  __syncthreads();
  if (tid == 0) {
    float m = fmaxf(fmaxf(wmaxs[0], wmaxs[1]), fmaxf(wmaxs[2], wmaxs[3]));
    int key = __float_as_int(m);
    key = key >= 0 ? key : (key ^ 0x7fffffff);            // monotonic float->int
    atomicMax(&kmax[bh], key);
  }
}

// ---------------- stage 3: qp = exp(dash - diag - rowmax) + eps ----------------
__global__ void __launch_bounds__(128)
k_qp(const _Float16* __restrict__ qh, const _Float16* __restrict__ Pt_g,
     const float* __restrict__ dq, _Float16* __restrict__ qp) {
  __shared__ alignas(16) _Float16 pt[4096];
  __shared__ float ldiag[64];
  int bh = blockIdx.x >> 4, tile = blockIdx.x & 15;
  int tid = threadIdx.x, w = tid >> 5, lane = tid & 31;
  {
    const uint4* s = (const uint4*)Pt_g; uint4* d = (uint4*)pt;
    for (int i = tid; i < 512; i += 128) d[i] = s[i];
  }
  int rowbase = bh * S_ + tile * 64;
  if (tid < 64) ldiag[tid] = dq[rowbase + tid];
  __syncthreads();
  v8f c[4];
  proj_tiles(qh + (size_t)(rowbase + w * 16) * 64, pt, c, lane);
  int row = lane & 15, half = lane >> 4;
  float mrow[8];
#pragma unroll
  for (int r = 0; r < 8; ++r) {
    mrow[r] = fmaxf(fmaxf(c[0][r], c[1][r]), fmaxf(c[2][r], c[3][r]));
#pragma unroll
    for (int off = 1; off < 16; off <<= 1)
      mrow[r] = fmaxf(mrow[r], __shfl_xor(mrow[r], off, 32));  // within 16-lane half
  }
  _Float16* ob = qp + (size_t)(rowbase + w * 16) * 64;
#pragma unroll
  for (int n = 0; n < 4; ++n)
#pragma unroll
    for (int r = 0; r < 8; ++r) {
      int rr = half * 8 + r;
      float val = __expf(c[n][r] - ldiag[w * 16 + rr] - mrow[r]) + EPSF;
      ob[rr * 64 + n * 16 + row] = (_Float16)val;
    }
}

// ---------------- stage 4: kp = exp(kdash - kmax) + eps ----------------
__global__ void k_kexp(const float* __restrict__ kdash, const int* __restrict__ kmax,
                       _Float16* __restrict__ kp) {
  size_t i = (size_t)blockIdx.x * 256 + threadIdx.x;
  int bh = (int)(i >> 16);                 // S_*M_ = 65536
  int key = kmax[bh];
  float m = key >= 0 ? __int_as_float(key) : __int_as_float(key ^ 0x7fffffff);
  kp[i] = (_Float16)(__expf(kdash[i] - m) + EPSF);
}

// ---------------- stage 5: causal flash over random features ----------------
// Double-buffered async (ASYNCcnt) staging of kp/vt tiles, WMMA compute.
__global__ void __launch_bounds__(128)
k_attn(const _Float16* __restrict__ qp, const _Float16* __restrict__ kp,
       const _Float16* __restrict__ vt, float* __restrict__ out) {
  __shared__ alignas(16) _Float16 lkp[2][64 * 64];   // [buf][key][feat]
  __shared__ alignas(16) _Float16 lvt[2][64 * 64];   // [buf][dim][key]
  __shared__ alignas(16) _Float16 ls[4][16 * 64];    // per-wave S strip [row][key]
  int bh = blockIdx.x >> 4, qtile = blockIdx.x & 15;
  int tid = threadIdx.x, w = tid >> 5, lane = tid & 31;
  int row = lane & 15, half = lane >> 4;

  const char* kp_bh = (const char*)(kp + (size_t)bh * S_ * 64);   // [key][feat]
  const char* vt_bh = (const char*)(vt + (size_t)bh * 64 * S_);   // [dim][s]
  unsigned lkp_off[2] = { (unsigned)(uintptr_t)&lkp[0][0], (unsigned)(uintptr_t)&lkp[1][0] };
  unsigned lvt_off[2] = { (unsigned)(uintptr_t)&lvt[0][0], (unsigned)(uintptr_t)&lvt[1][0] };

  // issue 8 async b128 copies per wave for key-tile j into buffer `buf`
  auto issue_tile = [&](int j, int buf) {
    {   // kp tile: 8 KB contiguous; wave w takes bytes [w*2048, +2048)
      const char* g = kp_bh + (size_t)j * 8192 + w * 2048 + lane * 16;
      unsigned    l = lkp_off[buf] + w * 2048 + lane * 16;
#pragma unroll
      for (int i = 0; i < 4; ++i) async_b128(l + i * 512, g + i * 512);
    }
    {   // vt tile: 64 rows x 128 B, global row stride 2048 B; wave w -> rows [w*16, +16)
      int r0 = w * 16 + (lane >> 3);
      const char* g = vt_bh + (size_t)r0 * 2048 + j * 128 + (lane & 7) * 16;
      unsigned    l = lvt_off[buf] + r0 * 128 + (lane & 7) * 16;
#pragma unroll
      for (int i = 0; i < 4; ++i) async_b128(l + i * 512, g + (size_t)i * 8192);
    }
  };

  const _Float16* qrow = qp + (size_t)(bh * S_ + qtile * 64 + w * 16) * 64;
  v16h aq0 = load_afrag(qrow, 64, row, 0, half);   // feats 0..31
  v16h aq1 = load_afrag(qrow, 64, row, 32, half);  // feats 32..63

  v8f o[4] = {};
  float rs[8];
#pragma unroll
  for (int r = 0; r < 8; ++r) rs[r] = 0.f;

  issue_tile(0, 0);

  for (int j = 0; j <= qtile; ++j) {
    int cur = j & 1;
    if (j < qtile) {
      issue_tile(j + 1, cur ^ 1);                       // prefetch next tile
      asm volatile("s_wait_asynccnt 0x8" ::: "memory"); // tile j done (in-order)
    } else {
      asm volatile("s_wait_asynccnt 0x0" ::: "memory");
    }
    __syncthreads();   // all waves' tile-j data visible in LDS

    const _Float16* ckp = lkp[cur];
    const _Float16* cvt = lvt[cur];

    // S = qp @ kp^T  (16 rows x 64 keys per wave)
    v8f sc[4];
#pragma unroll
    for (int n = 0; n < 4; ++n) {
      v8f acc = {};
      v16h b0 = load_bfrag(ckp, 64, n * 16 + row, 0, half);
      v16h b1 = load_bfrag(ckp, 64, n * 16 + row, 32, half);
      acc = wmma16(aq0, b0, acc);
      acc = wmma16(aq1, b1, acc);
      sc[n] = acc;
    }

    if (j == qtile) {   // causal mask on the diagonal tile: keep key <= query
#pragma unroll
      for (int n = 0; n < 4; ++n)
#pragma unroll
        for (int r = 0; r < 8; ++r) {
          int qr = w * 16 + half * 8 + r;
          int kc = n * 16 + row;
          sc[n][r] = (kc > qr) ? 0.0f : sc[n][r];
        }
    }

    // rowsum partials + re-layout S through this wave's LDS strip
    _Float16* myS = ls[w];
#pragma unroll
    for (int n = 0; n < 4; ++n)
#pragma unroll
      for (int r = 0; r < 8; ++r) {
        rs[r] += sc[n][r];
        myS[(half * 8 + r) * 64 + n * 16 + row] = (_Float16)sc[n][r];
      }
    // LDS ops from one wave are processed in order (DScnt); safe wave-local RAW
    v16h p0 = load_afrag(myS, 64, row, 0, half);   // keys 0..31
    v16h p1 = load_afrag(myS, 64, row, 32, half);  // keys 32..63

    // O += P @ V
#pragma unroll
    for (int n = 0; n < 4; ++n) {
      v16h b0 = load_bfrag(cvt, 64, n * 16 + row, 0, half);
      v16h b1 = load_bfrag(cvt, 64, n * 16 + row, 32, half);
      o[n] = wmma16(p0, b0, o[n]);
      o[n] = wmma16(p1, b1, o[n]);
    }
    __syncthreads();   // readers done before this buffer is refilled (j+2)
  }

  // full row sums: reduce across the 16 lanes sharing each row
#pragma unroll
  for (int r = 0; r < 8; ++r)
#pragma unroll
    for (int off = 1; off < 16; off <<= 1)
      rs[r] += __shfl_xor(rs[r], off, 32);

  float* ob = out + (size_t)(bh * S_ + qtile * 64 + w * 16) * 64;
#pragma unroll
  for (int n = 0; n < 4; ++n)
#pragma unroll
    for (int r = 0; r < 8; ++r)
      ob[(half * 8 + r) * 64 + n * 16 + row] = o[n][r] / rs[r];
}

// ---------------- host ----------------
extern "C" void kernel_launch(void* const* d_in, const int* in_sizes, int n_in,
                              void* d_out, int out_size, void* d_ws, size_t ws_size,
                              hipStream_t stream) {
  const float* q = (const float*)d_in[0];
  const float* k = (const float*)d_in[1];
  const float* v = (const float*)d_in[2];
  const float* P = (const float*)d_in[3];
  float* out = (float*)d_out;

  char* ws = (char*)d_ws;
  size_t o = 0;
  _Float16* qh    = (_Float16*)(ws + o); o += (size_t)BH_ * S_ * D_ * 2;   // 8 MB
  _Float16* kh    = (_Float16*)(ws + o); o += (size_t)BH_ * S_ * D_ * 2;
  _Float16* vt    = (_Float16*)(ws + o); o += (size_t)BH_ * S_ * D_ * 2;
  _Float16* qp    = (_Float16*)(ws + o); o += (size_t)BH_ * S_ * M_ * 2;
  _Float16* kp    = (_Float16*)(ws + o); o += (size_t)BH_ * S_ * M_ * 2;
  float*    kdash = (float*)(ws + o);    o += (size_t)BH_ * S_ * M_ * 4;   // 16 MB
  float*    dq    = (float*)(ws + o);    o += (size_t)BH_ * S_ * 4;
  float*    dk    = (float*)(ws + o);    o += (size_t)BH_ * S_ * 4;
  _Float16* Pt    = (_Float16*)(ws + o); o += (size_t)D_ * M_ * 2;
  int*      kmax  = (int*)(ws + o);      o += (size_t)BH_ * 4;

  k_prep_qk<<<(BH_ * S_ * D_) / 256, 256, 0, stream>>>(q, k, qh, kh);
  k_diag<<<(BH_ * S_ * 32) / 256, 256, 0, stream>>>(q, k, dq, dk);
  k_vt<<<BH_ * (S_ / 64), 256, 0, stream>>>(v, vt);
  k_pt<<<1, 256, 0, stream>>>(P, Pt, kmax);
  k_kdash<<<BH_ * (S_ / 64), 128, 0, stream>>>(kh, Pt, dk, kdash, kmax);
  k_qp<<<BH_ * (S_ / 64), 128, 0, stream>>>(qh, Pt, dq, qp);
  k_kexp<<<(BH_ * S_ * M_) / 256, 256, 0, stream>>>(kdash, kmax, kp);
  k_attn<<<BH_ * (S_ / 64), 128, 0, stream>>>(qp, kp, vt, out);
}
